// Encoder_9259949490960
// MI455X (gfx1250) — matrix operand order repaired and checked by reading, not compiled, not run on hardware
//
#include <hip/hip_runtime.h>
#include <stdint.h>

// ---------------- CDNA5 WMMA types ----------------
typedef __attribute__((ext_vector_type(16))) __bf16     v16bf;
typedef __attribute__((ext_vector_type(8)))  float      v8f;
typedef __attribute__((ext_vector_type(4)))  unsigned   u32x4;

#define DEV static __device__ __forceinline__

constexpr int B_  = 32;
constexpr int L_  = 512;
constexpr int H_  = 8;
constexpr int HD_ = 64;
constexpr int M_  = 512;
constexpr int F_  = 2048;
constexpr int N_  = B_ * L_;   // 16384 rows

// float -> bf16, round to nearest even (storage as raw u16)
DEV unsigned short f2bf(float f) {
  union { float f; unsigned u; } x; x.f = f;
  unsigned u = x.u;
  unsigned r = u + 0x7fffu + ((u >> 16) & 1u);
  return (unsigned short)(r >> 16);
}

union FragAB { v16bf v; u32x4 q[2]; };

// A fragment (16x32 bf16, MxK). Per ISA layout: lane = hi*16 + m;
// VGPR0..3 hold K = hi*8 + 0..7, VGPR4..7 hold K = 16 + hi*8 + 0..7.
DEV v16bf load_a(const unsigned short* A, int lda, int m0, int k0, int lane) {
  FragAB f;
  int m  = m0 + (lane & 15);
  int hi = lane >> 4;
  const unsigned short* p = A + (size_t)m * lda + k0 + hi * 8;
  f.q[0] = *(const u32x4*)(p);
  f.q[1] = *(const u32x4*)(p + 16);
  return f.v;
}

// B fragment (32x16 bf16, KxN) from a pre-transposed Bt (ncols x K row-major).
// Lane = kh*16 + n; lane holds K = kh*8 + 0..7 and K = 16 + kh*8 + 0..7.
DEV v16bf load_bt(const unsigned short* Bt, int ldb, int n0, int k0, int lane) {
  FragAB f;
  int n = n0 + (lane & 15);
  const unsigned short* p = Bt + (size_t)n * ldb + k0 + (lane >> 4) * 8;
  f.q[0] = *(const u32x4*)(p);
  f.q[1] = *(const u32x4*)(p + 16);
  return f.v;
}

DEV v8f wmma_bf16(v16bf a, v16bf b, v8f c) {
  // 8 args: neg_a, A, neg_b, B, c_mod, C, reuse_a, reuse_b
  return __builtin_amdgcn_wmma_f32_16x16x32_bf16(false, a, false, b, (short)0, c,
                                                 false, false);
}

// ---------------- prep kernels ----------------
__global__ void k_f32_to_bf16(const float* __restrict__ in,
                              unsigned short* __restrict__ out, int n) {
  int i = blockIdx.x * blockDim.x + threadIdx.x;
  if (i < n) out[i] = f2bf(in[i]);
}

// in: rows x cols (f32, row-major) -> out[c*rows + r] (bf16) == transpose
__global__ void k_transpose_bf16(const float* __restrict__ in,
                                 unsigned short* __restrict__ out,
                                 int rows, int cols) {
  int i = blockIdx.x * blockDim.x + threadIdx.x;
  if (i >= rows * cols) return;
  int r = i / cols, c = i % cols;
  out[(size_t)c * rows + r] = f2bf(in[i]);
}

// pre_linear_w (3,H,M,HD) -> qkvT[(g*HD + d)*M + m], g = t*H+h
__global__ void k_prep_qkvw(const float* __restrict__ in,
                            unsigned short* __restrict__ out) {
  int i = blockIdx.x * blockDim.x + threadIdx.x;
  if (i >= 3 * H_ * M_ * HD_) return;
  int d = i % HD_;
  int m = (i / HD_) % M_;
  int g = i / (M_ * HD_);
  out[((size_t)g * HD_ + d) * M_ + m] = f2bf(in[i]);
}

// ---------------- epilogue helpers ----------------
DEV void store_tile(v8f acc, int n0, int c0, int lane,
                    const float* __restrict__ bias,
                    const float* __restrict__ resid,
                    float* __restrict__ outF,
                    unsigned short* __restrict__ outB,
                    int ncols, int flags) {
  int c = c0 + (lane & 15);
  float bv = (flags & 1) ? bias[c] : 0.f;
  int rbase = n0 + ((lane >> 4) * 8);
  for (int r = 0; r < 8; ++r) {
    size_t idx = (size_t)(rbase + r) * ncols + c;
    float v = acc[r] + bv;
    if (flags & 2) v += resid[idx];
    if (flags & 4) v = fmaxf(v, 0.f);
    if (outF) outF[idx] = v;
    if (outB) outB[idx] = f2bf(v);
  }
}

DEV void qkv_store_tile(v8f acc, int n0, int c0, int lane,
                        const float* __restrict__ bias,
                        unsigned short* __restrict__ qbuf,
                        unsigned short* __restrict__ kbuf,
                        unsigned short* __restrict__ vtbuf) {
  int c = c0 + (lane & 15);
  int t = c / (H_ * HD_);
  int h = (c / HD_) % H_;
  int d = c % HD_;
  float bv = bias[(t * H_ + h) * HD_ + d];
  int rbase = n0 + ((lane >> 4) * 8);
  for (int r = 0; r < 8; ++r) {
    int n = rbase + r;
    unsigned short bf = f2bf(acc[r] + bv);
    if (t == 0) {
      qbuf[((size_t)h * N_ + n) * HD_ + d] = bf;
    } else if (t == 1) {
      kbuf[((size_t)h * N_ + n) * HD_ + d] = bf;
    } else {
      int b = n / L_, l = n % L_;
      vtbuf[(((size_t)h * B_ + b) * HD_ + d) * L_ + l] = bf;
    }
  }
}

// ---------------- fused QKV projection GEMM (32x32 per wave) ----------------
// C(N x 1536) = A(N x 512) @ WtT ; epilogue scatters into q/k (H,N,HD), vT (H,B,HD,L)
// block: 8 waves = 4 (rows) x 2 (cols) -> 128 x 64 block tile
__global__ void __launch_bounds__(256)
k_qkv_gemm(const unsigned short* __restrict__ A,
           const unsigned short* __restrict__ Wt,
           const float* __restrict__ bias,
           unsigned short* __restrict__ qbuf,
           unsigned short* __restrict__ kbuf,
           unsigned short* __restrict__ vtbuf) {
  int lane = threadIdx.x & 31;
  int wave = threadIdx.x >> 5;
  int n0 = blockIdx.y * 128 + (wave >> 1) * 32;
  int c0 = blockIdx.x * 64 + (wave & 1) * 32;
  v8f acc00 = {}, acc01 = {}, acc10 = {}, acc11 = {};
#pragma unroll 1
  for (int k = 0; k < M_; k += 32) {
    v16bf a0 = load_a(A, M_, n0, k, lane);
    v16bf a1 = load_a(A, M_, n0 + 16, k, lane);
    v16bf b0 = load_bt(Wt, M_, c0, k, lane);
    v16bf b1 = load_bt(Wt, M_, c0 + 16, k, lane);
    acc00 = wmma_bf16(a0, b0, acc00);
    acc01 = wmma_bf16(a0, b1, acc01);
    acc10 = wmma_bf16(a1, b0, acc10);
    acc11 = wmma_bf16(a1, b1, acc11);
  }
  qkv_store_tile(acc00, n0,      c0,      lane, bias, qbuf, kbuf, vtbuf);
  qkv_store_tile(acc01, n0,      c0 + 16, lane, bias, qbuf, kbuf, vtbuf);
  qkv_store_tile(acc10, n0 + 16, c0,      lane, bias, qbuf, kbuf, vtbuf);
  qkv_store_tile(acc11, n0 + 16, c0 + 16, lane, bias, qbuf, kbuf, vtbuf);
}

// ---------------- fused attention ----------------
// block = 8 waves, handles one (h,b) and 32 query rows.
// LDS: 32x512 f32 scores + 32x512 bf16 probs = 96 KB dynamic.
__global__ void __launch_bounds__(256)
k_attention(const unsigned short* __restrict__ qbuf,
            const unsigned short* __restrict__ kbuf,
            const unsigned short* __restrict__ vtbuf,
            const float* __restrict__ mask,
            unsigned short* __restrict__ ctx) {
  extern __shared__ char smem[];
  float* sS = (float*)smem;                                    // 32 x 512 f32
  unsigned short* sP = (unsigned short*)(smem + 32 * L_ * 4);  // 32 x 512 bf16

  int hb = blockIdx.x;            // h*B + b
  int h = hb / B_, b = hb % B_;
  int q0 = blockIdx.y * 32;
  int lane = threadIdx.x & 31;
  int wave = threadIdx.x >> 5;

  const unsigned short* Qb = qbuf + (size_t)hb * L_ * HD_;
  const unsigned short* Kb = kbuf + (size_t)hb * L_ * HD_;
  const unsigned short* Vt = vtbuf + (size_t)hb * HD_ * L_;

  // phase 1: scores = Q @ K^T (+mask) -> LDS. Q fragments hoisted & reused.
  {
    int qt = wave >> 2;            // 0..1   (16-row q tile)
    int kseg = (wave & 3) * 128;   // 128 keys per wave
    v16bf a_lo = load_a(Qb, HD_, q0 + qt * 16, 0, lane);
    v16bf a_hi = load_a(Qb, HD_, q0 + qt * 16, 32, lane);
#pragma unroll 1
    for (int kt = 0; kt < 8; ++kt) {
      int key0 = kseg + kt * 16;
      v8f acc = {};
      acc = wmma_bf16(a_lo, load_bt(Kb, HD_, key0, 0, lane), acc);
      acc = wmma_bf16(a_hi, load_bt(Kb, HD_, key0, 32, lane), acc);
      int col = key0 + (lane & 15);
      float mv = mask[b * L_ + col];
      int rbase = qt * 16 + ((lane >> 4) * 8);
      for (int r = 0; r < 8; ++r)
        sS[(rbase + r) * L_ + col] = acc[r] + mv;
    }
  }
  __syncthreads();

  // phase 2: softmax over each of the 32 rows (8 threads / row)
  {
    int row = threadIdx.x >> 3;
    int seg = threadIdx.x & 7;
    float* S = sS + row * L_ + seg * 64;
    float m = -3.0e38f;
    for (int i = 0; i < 64; ++i) m = fmaxf(m, S[i]);
    for (int o = 4; o >= 1; o >>= 1) m = fmaxf(m, __shfl_xor(m, o, 8));
    float s = 0.f;
    for (int i = 0; i < 64; ++i) { float e = __expf(S[i] - m); S[i] = e; s += e; }
    for (int o = 4; o >= 1; o >>= 1) s += __shfl_xor(s, o, 8);
    float inv = 1.f / s;
    unsigned short* P = sP + row * L_ + seg * 64;
    for (int i = 0; i < 64; ++i) P[i] = f2bf(S[i] * inv);
  }
  __syncthreads();

  // phase 3: ctx = P @ V   (A from LDS, B = V^T from global)
  {
    int qt = wave >> 2;      // 16-row q tile
    int dt = wave & 3;       // 16-col d tile (HD = 64)
    v8f acc = {};
#pragma unroll 1
    for (int k = 0; k < L_; k += 32) {
      v16bf a  = load_a(sP, L_, qt * 16, k, lane);
      v16bf bb = load_bt(Vt, L_, dt * 16, k, lane);
      acc = wmma_bf16(a, bb, acc);
    }
    int col = h * HD_ + dt * 16 + (lane & 15);
    int lrow = q0 + qt * 16 + ((lane >> 4) * 8);
    for (int r = 0; r < 8; ++r) {
      size_t n = (size_t)b * L_ + lrow + r;
      ctx[n * M_ + col] = f2bf(acc[r]);
    }
  }
}

// ---------------- generic bf16 GEMM, 32x32 per wave, fused epilogue ---------
// flags: 1 = +bias[col], 2 = +resid[idx] (f32), 4 = relu
// block: 8 waves = 4 (rows) x 2 (cols) -> 128 x 64 block tile
__global__ void __launch_bounds__(256)
k_gemm(const unsigned short* __restrict__ A,
       const unsigned short* __restrict__ Bt,
       const float* __restrict__ bias,
       const float* __restrict__ resid,
       float* __restrict__ outF,
       unsigned short* __restrict__ outB,
       int K, int ncols, int flags) {
  int lane = threadIdx.x & 31;
  int wave = threadIdx.x >> 5;
  int n0 = blockIdx.y * 128 + (wave >> 1) * 32;
  int c0 = blockIdx.x * 64 + (wave & 1) * 32;
  v8f acc00 = {}, acc01 = {}, acc10 = {}, acc11 = {};
#pragma unroll 1
  for (int k = 0; k < K; k += 32) {
    v16bf a0 = load_a(A, K, n0, k, lane);
    v16bf a1 = load_a(A, K, n0 + 16, k, lane);
    v16bf b0 = load_bt(Bt, K, c0, k, lane);
    v16bf b1 = load_bt(Bt, K, c0 + 16, k, lane);
    acc00 = wmma_bf16(a0, b0, acc00);
    acc01 = wmma_bf16(a0, b1, acc01);
    acc10 = wmma_bf16(a1, b0, acc10);
    acc11 = wmma_bf16(a1, b1, acc11);
  }
  store_tile(acc00, n0,      c0,      lane, bias, resid, outF, outB, ncols, flags);
  store_tile(acc01, n0,      c0 + 16, lane, bias, resid, outF, outB, ncols, flags);
  store_tile(acc10, n0 + 16, c0,      lane, bias, resid, outF, outB, ncols, flags);
  store_tile(acc11, n0 + 16, c0 + 16, lane, bias, resid, outF, outB, ncols, flags);
}

// ---------------- layernorm (one wave per 512-elem row, 8 rows/block) --------
__global__ void __launch_bounds__(256)
k_layernorm(const float* __restrict__ x,
            float* __restrict__ outF,
            unsigned short* __restrict__ outB,
            const float* __restrict__ w,
            const float* __restrict__ bvec) {
  int lane = threadIdx.x & 31;
  int wave = threadIdx.x >> 5;
  size_t row = (size_t)blockIdx.x * 8 + wave;
  const float* xr = x + row * M_;
  float v[16];
  float s = 0.f;
  for (int i = 0; i < 16; ++i) { v[i] = xr[lane + i * 32]; s += v[i]; }
  for (int o = 16; o >= 1; o >>= 1) s += __shfl_xor(s, o, 32);
  float mu = s * (1.f / M_);
  float q = 0.f;
  for (int i = 0; i < 16; ++i) { float d = v[i] - mu; q += d * d; }
  for (int o = 16; o >= 1; o >>= 1) q += __shfl_xor(q, o, 32);
  float rstd = rsqrtf(q * (1.f / M_) + 1e-5f);
  for (int i = 0; i < 16; ++i) {
    int cidx = lane + i * 32;
    float y = (v[i] - mu) * rstd * w[cidx] + bvec[cidx];
    outF[row * M_ + cidx] = y;
    if (outB) outB[row * M_ + cidx] = f2bf(y);
  }
}

// ---------------- host side ----------------
extern "C" void kernel_launch(void* const* d_in, const int* in_sizes, int n_in,
                              void* d_out, int out_size, void* d_ws, size_t ws_size,
                              hipStream_t stream) {
  (void)in_sizes; (void)n_in; (void)out_size; (void)ws_size;
  const float* inp   = (const float*)d_in[0];
  const float* mask  = (const float*)d_in[1];
  const float* preW  = (const float*)d_in[2];
  const float* preB  = (const float*)d_in[3];
  const float* postW = (const float*)d_in[4];
  const float* postB = (const float*)d_in[5];
  const float* ff1W  = (const float*)d_in[6];
  const float* ff1B  = (const float*)d_in[7];
  const float* ff2W  = (const float*)d_in[8];
  const float* ff2B  = (const float*)d_in[9];
  const float* ln1w  = (const float*)d_in[10];
  const float* ln1b  = (const float*)d_in[11];
  const float* ln2w  = (const float*)d_in[12];
  const float* ln2b  = (const float*)d_in[13];
  float* out = (float*)d_out;

  char* ws = (char*)d_ws;
  size_t off = 0;
  auto alloc = [&](size_t bytes) -> char* {
    char* p = ws + off;
    off = (off + bytes + 255) & ~(size_t)255;
    return p;
  };
  unsigned short* qkvT  = (unsigned short*)alloc((size_t)3 * H_ * HD_ * M_ * 2);
  unsigned short* postT = (unsigned short*)alloc((size_t)M_ * M_ * 2);
  unsigned short* ff1T  = (unsigned short*)alloc((size_t)F_ * M_ * 2);
  unsigned short* ff2T  = (unsigned short*)alloc((size_t)M_ * F_ * 2);
  unsigned short* inpB  = (unsigned short*)alloc((size_t)N_ * M_ * 2);      // reused as ctx
  unsigned short* qbuf  = (unsigned short*)alloc((size_t)H_ * N_ * HD_ * 2); // reused as saB
  unsigned short* kbuf  = (unsigned short*)alloc((size_t)H_ * N_ * HD_ * 2);
  unsigned short* vtbuf = (unsigned short*)alloc((size_t)H_ * N_ * HD_ * 2);
  float*          saF   = (float*)alloc((size_t)N_ * M_ * 4);
  unsigned short* ffB   = (unsigned short*)alloc((size_t)N_ * F_ * 2);
  unsigned short* ctx = inpB;  // inp_bf16 dead after QKV GEMM
  unsigned short* saB = qbuf;  // q dead after attention

  // weight / activation conversions (bf16, transposed for contiguous B frags)
  { int n = N_ * M_;           k_f32_to_bf16  <<<(n + 255) / 256, 256, 0, stream>>>(inp, inpB, n); }
  { int n = 3 * H_ * M_ * HD_; k_prep_qkvw    <<<(n + 255) / 256, 256, 0, stream>>>(preW, qkvT); }
  { int n = M_ * M_;           k_transpose_bf16<<<(n + 255) / 256, 256, 0, stream>>>(postW, postT, M_, M_); }
  { int n = M_ * F_;           k_transpose_bf16<<<(n + 255) / 256, 256, 0, stream>>>(ff1W, ff1T, M_, F_); }
  { int n = F_ * M_;           k_transpose_bf16<<<(n + 255) / 256, 256, 0, stream>>>(ff2W, ff2T, F_, M_); }

  // fused QKV projection: 1536 cols / 64 per block, N / 128 rows per block
  k_qkv_gemm<<<dim3((3 * H_ * HD_) / 64, N_ / 128), 256, 0, stream>>>(
      inpB, qkvT, preB, qbuf, kbuf, vtbuf);

  // fused attention (scores -> softmax -> P@V), 96KB dynamic LDS
  k_attention<<<dim3(H_ * B_, L_ / 32), 256, 32 * L_ * 4 + 32 * L_ * 2, stream>>>(
      qbuf, kbuf, vtbuf, mask, ctx);

  // post linear + bias + residual(inp)
  k_gemm<<<dim3(M_ / 64, N_ / 128), 256, 0, stream>>>(
      ctx, postT, postB, inp, saF, nullptr, M_, M_, 1 | 2);

  // LN1 (in place, also emit bf16 for FF1)
  k_layernorm<<<N_ / 8, 256, 0, stream>>>(saF, saF, saB, ln1w, ln1b);

  // FF1 + bias + relu (bf16 out only)
  k_gemm<<<dim3(F_ / 64, N_ / 128), 256, 0, stream>>>(
      saB, ff1T, ff1B, nullptr, nullptr, ffB, M_, F_, 1 | 4);

  // FF2 + bias + residual(sa)  -> d_out (f32)
  k_gemm<<<dim3(M_ / 64, N_ / 128), 256, 0, stream>>>(
      ffB, ff2T, ff2B, saF, out, nullptr, F_, M_, 1 | 2);

  // LN2 in place on d_out
  k_layernorm<<<N_ / 8, 256, 0, stream>>>(out, out, nullptr, ln2w, ln2b);
}
